// TransformerBlock_11046655885339
// MI455X (gfx1250) — matrix operand order repaired and checked by reading, not compiled, not run on hardware
//
#include <hip/hip_runtime.h>
#include <hip/hip_bf16.h>
#include <math.h>

#ifndef __has_builtin
#define __has_builtin(x) 0
#endif

typedef _Float16 f16;
typedef __attribute__((ext_vector_type(16))) _Float16 v16h;
typedef __attribute__((ext_vector_type(8)))  _Float16 v8h;
typedef __attribute__((ext_vector_type(8)))  float    v8f;
typedef int v4i_ __attribute__((vector_size(16)));   // matches builtin param type

#define B_  8
#define S_  1024
#define D_  768
#define F_  3072
#define H_  12
#define DK_ 64
#define MS_ (B_*S_)   // 8192 rows

// ---------------------------------------------------------------------------
// WMMA helpers (gfx1250, wave32). Layouts used by this code (A/B K order is its own choice):
//  A (16x32 f16): lane&15 = M; frag elem e -> K = (lane>>4)*16 + e   (contiguous)
//  B (32x16 f16): lane&15 = N; frag elem e -> K = (lane>>4)*16 + e   (contiguous)
//  C/D (16x16 f32): lane&15 = N; slot r -> M = (lane>>4)*8 + r
// ---------------------------------------------------------------------------
__device__ __forceinline__ v8f wmma_f16(v16h a, v16h b, v8f c) {
  return __builtin_amdgcn_wmma_f32_16x16x32_f16(false, a, false, b, (short)0, c,
                                                false, false);
}

__device__ __forceinline__ v16h make_afrag(const f16* rowp, int half_id) {
  v8h lo = *(const v8h*)(rowp + half_id * 16);
  v8h hi = *(const v8h*)(rowp + half_id * 16 + 8);
  v16h r;
#pragma unroll
  for (int e = 0; e < 8; ++e) { r[e] = lo[e]; r[8 + e] = hi[e]; }
  return r;
}

// ---------------------------------------------------------------------------
// Async global->LDS copy (CDNA5 GLOBAL_LOAD_ASYNC_TO_LDS_B128, ASYNCcnt).
// Builtin signature (from hipcc diagnostic): (v4i as1*, v4i as3*, Ii, Ii).
// Guarded: falls back to a synchronous bounce copy if the builtin is absent.
// ---------------------------------------------------------------------------
#if __has_builtin(__builtin_amdgcn_global_load_async_to_lds_b128)
#define ASYNC_CP 1
#else
#define ASYNC_CP 0
#endif

__device__ __forceinline__ void cp16_async(f16* l, const f16* g) {
#if ASYNC_CP
  __builtin_amdgcn_global_load_async_to_lds_b128(
      (__attribute__((address_space(1))) v4i_*)g,
      (__attribute__((address_space(3))) v4i_*)l, 0, 0);
#else
  *(v8h*)l = *(const v8h*)g;
#endif
}

// Wait until at most N async copies are outstanding (drains the older stage).
__device__ __forceinline__ void wait_cp_prev() {
#if ASYNC_CP
#if __has_builtin(__builtin_amdgcn_s_wait_asynccnt)
  __builtin_amdgcn_s_wait_asynccnt(5);
#else
  asm volatile("s_wait_asynccnt 0x5" ::: "memory");
#endif
#endif
}
__device__ __forceinline__ void wait_cp_all() {
#if ASYNC_CP
#if __has_builtin(__builtin_amdgcn_s_wait_asynccnt)
  __builtin_amdgcn_s_wait_asynccnt(0);
#else
  asm volatile("s_wait_asynccnt 0x0" ::: "memory");
#endif
#endif
}

// ---------------------------------------------------------------------------
// Elementwise prep kernels
// ---------------------------------------------------------------------------
__global__ void k_cast_f16(const float* __restrict__ in, f16* __restrict__ out, int n) {
  int i = blockIdx.x * 256 + threadIdx.x;
  if (i < n) out[i] = (f16)in[i];
}

// W [K,N] f32 -> Wt [N,K] f16
__global__ void k_wtrans(const float* __restrict__ W, f16* __restrict__ Wt, int K, int N) {
  int i = blockIdx.x * 256 + threadIdx.x;
  if (i < K * N) {
    int k = i / N, n = i - k * N;
    Wt[(size_t)n * K + k] = (f16)W[i];
  }
}

// V f16 [B*S, D] -> Vt f16 [(B*H*DK), S]  (per-head feature-major)
__global__ void k_vtrans(const f16* __restrict__ V, f16* __restrict__ Vt) {
  int i = blockIdx.x * 256 + threadIdx.x;
  if (i < MS_ * D_) {
    int bs = i / D_, f = i - bs * D_;
    int b = bs / S_, s = bs - b * S_;
    int h = f / DK_, d = f - h * DK_;
    Vt[(((size_t)b * H_ + h) * DK_ + d) * S_ + s] = V[i];
  }
}

// ---------------------------------------------------------------------------
// GEMM: C[M,N] = A[M,K] @ Bt[N,K]^T + bias, optional tanh-GELU, f32/f16 out.
// 256 threads = 8 waves; block tile 256x64; wave tile 32x64; K-step 32.
// Double-buffered LDS staged with async global->LDS copies (5 per thread per
// stage); s_wait_asynccnt<=5 drains only the previous stage so HBM latency
// overlaps WMMA issue.
// ---------------------------------------------------------------------------
__global__ __launch_bounds__(256) void k_gemm(const f16* __restrict__ A,
                                              const f16* __restrict__ Bt,
                                              const float* __restrict__ bias,
                                              float* __restrict__ Cf,
                                              f16* __restrict__ Ch,
                                              int M, int N, int K, int act_gelu) {
  __shared__ __align__(32) f16 sA[2][256 * 48];
  __shared__ __align__(32) f16 sB[2][64 * 48];
  const int tid = threadIdx.x;
  const int lane = tid & 31, wave = tid >> 5;
  const int half_id = lane >> 4, l16 = lane & 15;
  const int bm = blockIdx.x * 256, bn = blockIdx.y * 64;

  // Stage one K-step (32 cols) of A (256 rows) and B (64 rows) into buffer `buf`.
  auto stage = [&](int buf, int k0) {
#pragma unroll
    for (int it = 0; it < 4; ++it) {
      int idx = tid * 8 + it * 2048;
      int r = idx >> 5, c = idx & 31;
      cp16_async(&sA[buf][r * 48 + c], A + (size_t)(bm + r) * K + k0 + c);
    }
    {
      int idx = tid * 8;
      int r = idx >> 5, c = idx & 31;
      cp16_async(&sB[buf][r * 48 + c], Bt + (size_t)(bn + r) * K + k0 + c);
    }
  };

  v8f acc0[4] = {}, acc1[4] = {};
  const int nk = K / 32;
  stage(0, 0);
  for (int ik = 0; ik < nk; ++ik) {
    const int cur = ik & 1;
    if (ik + 1 < nk) {
      stage(cur ^ 1, (ik + 1) * 32);
      wait_cp_prev();   // previous stage's 5 copies complete (in-order counter)
    } else {
      wait_cp_all();
    }
    __syncthreads();

    const f16* bufA = sA[cur];
    const f16* bufB = sB[cur];
    v16h a0 = make_afrag(bufA + (wave * 32 + l16) * 48, half_id);
    v16h a1 = make_afrag(bufA + (wave * 32 + 16 + l16) * 48, half_id);
#pragma unroll
    for (int nc = 0; nc < 4; ++nc) {
      v16h bf = *(const v16h*)(bufB + (nc * 16 + l16) * 48 + half_id * 16);
      acc0[nc] = wmma_f16(a0, bf, acc0[nc]);
      acc1[nc] = wmma_f16(a1, bf, acc1[nc]);
    }
    __syncthreads();   // everyone done reading `cur` before it is re-staged
  }

  // Epilogue: bias, optional GELU(tanh), scattered stores.
#pragma unroll
  for (int g = 0; g < 2; ++g) {
#pragma unroll
    for (int nc = 0; nc < 4; ++nc) {
      int col = bn + nc * 16 + l16;
      float bb = bias ? bias[col] : 0.0f;
#pragma unroll
      for (int r = 0; r < 8; ++r) {
        int row = bm + wave * 32 + g * 16 + half_id * 8 + r;
        float v = (g == 0 ? acc0[nc][r] : acc1[nc][r]) + bb;
        if (act_gelu) {
          float xx = v;
          float inner = 0.79788456080286536f * (xx + 0.044715f * xx * xx * xx);
          v = 0.5f * xx * (1.0f + tanhf(inner));
        }
        if (Cf) Cf[(size_t)row * N + col] = v;
        if (Ch) Ch[(size_t)row * N + col] = (f16)v;
      }
    }
  }
}

// ---------------------------------------------------------------------------
// Fused flash attention. 128 threads = 4 waves; wave handles 16 query rows.
// Loop over 32-key blocks: QK^T (4 WMMA), online softmax, PV (4 WMMA).
// ---------------------------------------------------------------------------
__global__ __launch_bounds__(128) void k_attn(const f16* __restrict__ Q,
                                              const f16* __restrict__ Kd,
                                              const f16* __restrict__ Vt,
                                              const int* __restrict__ mask,
                                              f16* __restrict__ O) {
  __shared__ __align__(32) f16 sP[4][16 * 48];
  const int tid = threadIdx.x;
  const int lane = tid & 31, wave = tid >> 5;
  const int half_id = lane >> 4, l16 = lane & 15;
  const int h = blockIdx.y, b = blockIdx.z;
  const int q0 = blockIdx.x * 64 + wave * 16;

  // Q fragments (held in registers), pre-scaled by 1/sqrt(d_k) = 0.125.
  const f16* qp = Q + (size_t)(b * S_ + q0 + l16) * D_ + h * DK_;
  v16h aQ0 = make_afrag(qp, half_id);
  v16h aQ1 = make_afrag(qp + 32, half_id);
#pragma unroll
  for (int e = 0; e < 16; ++e) {
    aQ0[e] *= (f16)0.125f;
    aQ1[e] *= (f16)0.125f;
  }

  float mrow[8], lrow[8], alpha[8];
  v8f o[4] = {};
#pragma unroll
  for (int r = 0; r < 8; ++r) { mrow[r] = -INFINITY; lrow[r] = 0.0f; }
  const int* mp = mask + b * S_;

  for (int j = 0; j < S_; j += 32) {
    v8f s0 = {}, s1 = {};
    {
      const f16* kp = Kd + (size_t)(b * S_ + j + l16) * D_ + h * DK_;
      v16h bk0 = *(const v16h*)(kp + half_id * 16);
      v16h bk1 = *(const v16h*)(kp + 32 + half_id * 16);
      s0 = wmma_f16(aQ0, bk0, s0);
      s0 = wmma_f16(aQ1, bk1, s0);
    }
    {
      const f16* kp = Kd + (size_t)(b * S_ + j + 16 + l16) * D_ + h * DK_;
      v16h bk0 = *(const v16h*)(kp + half_id * 16);
      v16h bk1 = *(const v16h*)(kp + 32 + half_id * 16);
      s1 = wmma_f16(aQ0, bk0, s1);
      s1 = wmma_f16(aQ1, bk1, s1);
    }
    if (mp[j + l16] == 0) {
#pragma unroll
      for (int r = 0; r < 8; ++r) s0[r] = -1e9f;
    }
    if (mp[j + 16 + l16] == 0) {
#pragma unroll
      for (int r = 0; r < 8; ++r) s1[r] = -1e9f;
    }
    // Online softmax: per-row max/sum via shfl_xor inside 16-lane groups.
#pragma unroll
    for (int r = 0; r < 8; ++r) {
      float mx = fmaxf(s0[r], s1[r]);
      mx = fmaxf(mx, __shfl_xor(mx, 1));
      mx = fmaxf(mx, __shfl_xor(mx, 2));
      mx = fmaxf(mx, __shfl_xor(mx, 4));
      mx = fmaxf(mx, __shfl_xor(mx, 8));
      float mn = fmaxf(mrow[r], mx);
      float a = __expf(mrow[r] - mn);
      mrow[r] = mn;
      float p0 = __expf(s0[r] - mn);
      float p1 = __expf(s1[r] - mn);
      s0[r] = p0;
      s1[r] = p1;
      float ps = p0 + p1;
      ps += __shfl_xor(ps, 1);
      ps += __shfl_xor(ps, 2);
      ps += __shfl_xor(ps, 4);
      ps += __shfl_xor(ps, 8);
      lrow[r] = lrow[r] * a + ps;
      alpha[r] = a;
    }
    // C-layout P -> A-layout fragment via per-wave LDS tile.
    f16* pw = sP[wave];
#pragma unroll
    for (int r = 0; r < 8; ++r) {
      int mr = half_id * 8 + r;
      pw[mr * 48 + l16] = (f16)s0[r];
      pw[mr * 48 + 16 + l16] = (f16)s1[r];
    }
    __syncthreads();
    v16h aP = make_afrag(sP[wave] + l16 * 48, half_id);
    __syncthreads();
    // Rescale running output, accumulate P @ V (V pre-transposed: contiguous).
    const f16* vb = Vt + ((size_t)(b * H_ + h) * DK_) * S_ + j;
#pragma unroll
    for (int nc = 0; nc < 4; ++nc) {
#pragma unroll
      for (int r = 0; r < 8; ++r) o[nc][r] *= alpha[r];
      v16h bv = *(const v16h*)(vb + (size_t)(nc * 16 + l16) * S_ + half_id * 16);
      o[nc] = wmma_f16(aP, bv, o[nc]);
    }
  }

  float invl[8];
#pragma unroll
  for (int r = 0; r < 8; ++r) invl[r] = 1.0f / lrow[r];
#pragma unroll
  for (int nc = 0; nc < 4; ++nc) {
#pragma unroll
    for (int r = 0; r < 8; ++r) {
      size_t row = (size_t)(b * S_ + q0 + half_id * 8 + r);
      O[row * D_ + h * DK_ + nc * 16 + l16] = (f16)(o[nc][r] * invl[r]);
    }
  }
}

// ---------------------------------------------------------------------------
// Residual add + LayerNorm (torch-style: unbiased var, eps added to STD).
// ---------------------------------------------------------------------------
__global__ __launch_bounds__(256) void k_add_ln(const float* __restrict__ X,
                                                const float* __restrict__ Y,
                                                const float* __restrict__ ga,
                                                const float* __restrict__ gb,
                                                float* __restrict__ out,
                                                f16* __restrict__ out16) {
  __shared__ float red[256];
  const int row = blockIdx.x, t = threadIdx.x;
  const float* xr = X + (size_t)row * D_;
  const float* yr = Y + (size_t)row * D_;
  float v[3];
#pragma unroll
  for (int j = 0; j < 3; ++j) v[j] = xr[t + j * 256] + yr[t + j * 256];

  red[t] = v[0] + v[1] + v[2];
  __syncthreads();
  for (int off = 128; off > 0; off >>= 1) {
    if (t < off) red[t] += red[t + off];
    __syncthreads();
  }
  float mean = red[0] * (1.0f / D_);
  __syncthreads();

  float ss = 0.0f;
#pragma unroll
  for (int j = 0; j < 3; ++j) { float d = v[j] - mean; ss += d * d; }
  red[t] = ss;
  __syncthreads();
  for (int off = 128; off > 0; off >>= 1) {
    if (t < off) red[t] += red[t + off];
    __syncthreads();
  }
  float var = red[0] / (float)(D_ - 1);
  float inv = 1.0f / (sqrtf(var) + 1e-6f);
#pragma unroll
  for (int j = 0; j < 3; ++j) {
    int c = t + j * 256;
    float o = ga[c] * (v[j] - mean) * inv + gb[c];
    out[(size_t)row * D_ + c] = o;
    if (out16) out16[(size_t)row * D_ + c] = (f16)o;
  }
}

// ---------------------------------------------------------------------------
// Host-side orchestration
// ---------------------------------------------------------------------------
extern "C" void kernel_launch(void* const* d_in, const int* in_sizes, int n_in,
                              void* d_out, int out_size, void* d_ws, size_t ws_size,
                              hipStream_t stream) {
  (void)in_sizes; (void)n_in; (void)out_size; (void)ws_size;
  const float* x    = (const float*)d_in[0];
  const int*   mask = (const int*)d_in[1];
  const float* Wq = (const float*)d_in[2];
  const float* bq = (const float*)d_in[3];
  const float* Wk = (const float*)d_in[4];
  const float* bk = (const float*)d_in[5];
  const float* Wv = (const float*)d_in[6];
  const float* bv = (const float*)d_in[7];
  const float* Wo = (const float*)d_in[8];
  const float* bo = (const float*)d_in[9];
  const float* W1 = (const float*)d_in[10];
  const float* b1 = (const float*)d_in[11];
  const float* W2 = (const float*)d_in[12];
  const float* b2 = (const float*)d_in[13];
  const float* ln1a = (const float*)d_in[14];
  const float* ln1b = (const float*)d_in[15];
  const float* ln2a = (const float*)d_in[16];
  const float* ln2b = (const float*)d_in[17];
  float* out = (float*)d_out;

  char* ws = (char*)d_ws;
  size_t off = 0;
  auto alloc = [&](size_t bytes) -> void* {
    void* p = ws + off;
    off = (off + bytes + 255) & ~(size_t)255;
    return p;
  };
  f16* x16  = (f16*)alloc((size_t)MS_ * D_ * 2);
  f16* q16  = (f16*)alloc((size_t)MS_ * D_ * 2);
  f16* k16  = (f16*)alloc((size_t)MS_ * D_ * 2);
  f16* v16b = (f16*)alloc((size_t)MS_ * D_ * 2);
  f16* vt16 = (f16*)alloc((size_t)MS_ * D_ * 2);
  f16* at16 = (f16*)alloc((size_t)MS_ * D_ * 2);
  f16* o116 = (f16*)alloc((size_t)MS_ * D_ * 2);
  f16* h116 = (f16*)alloc((size_t)MS_ * F_ * 2);
  f16* wqt  = (f16*)alloc((size_t)D_ * D_ * 2);
  f16* wkt  = (f16*)alloc((size_t)D_ * D_ * 2);
  f16* wvt  = (f16*)alloc((size_t)D_ * D_ * 2);
  f16* wot  = (f16*)alloc((size_t)D_ * D_ * 2);
  f16* w1t  = (f16*)alloc((size_t)D_ * F_ * 2);
  f16* w2t  = (f16*)alloc((size_t)F_ * D_ * 2);
  float* projf = (float*)alloc((size_t)MS_ * D_ * 4);
  float* out1f = (float*)alloc((size_t)MS_ * D_ * 4);
  float* ffnf  = (float*)alloc((size_t)MS_ * D_ * 4);

  const int nxd = MS_ * D_;
  k_cast_f16<<<nxd / 256, 256, 0, stream>>>(x, x16, nxd);
  k_wtrans<<<(D_ * D_) / 256, 256, 0, stream>>>(Wq, wqt, D_, D_);
  k_wtrans<<<(D_ * D_) / 256, 256, 0, stream>>>(Wk, wkt, D_, D_);
  k_wtrans<<<(D_ * D_) / 256, 256, 0, stream>>>(Wv, wvt, D_, D_);
  k_wtrans<<<(D_ * D_) / 256, 256, 0, stream>>>(Wo, wot, D_, D_);
  k_wtrans<<<(D_ * F_) / 256, 256, 0, stream>>>(W1, w1t, D_, F_);
  k_wtrans<<<(F_ * D_) / 256, 256, 0, stream>>>(W2, w2t, F_, D_);

  dim3 g1(MS_ / 256, D_ / 64);
  k_gemm<<<g1, 256, 0, stream>>>(x16, wqt, bq, nullptr, q16, MS_, D_, D_, 0);
  k_gemm<<<g1, 256, 0, stream>>>(x16, wkt, bk, nullptr, k16, MS_, D_, D_, 0);
  k_gemm<<<g1, 256, 0, stream>>>(x16, wvt, bv, nullptr, v16b, MS_, D_, D_, 0);
  k_vtrans<<<nxd / 256, 256, 0, stream>>>(v16b, vt16);

  dim3 ga(S_ / 64, H_, B_);
  k_attn<<<ga, 128, 0, stream>>>(q16, k16, vt16, mask, at16);

  k_gemm<<<g1, 256, 0, stream>>>(at16, wot, bo, projf, nullptr, MS_, D_, D_, 0);
  k_add_ln<<<MS_, 256, 0, stream>>>(x, projf, ln1a, ln1b, out1f, o116);

  dim3 g2(MS_ / 256, F_ / 64);
  k_gemm<<<g2, 256, 0, stream>>>(o116, w1t, b1, nullptr, h116, MS_, F_, D_, 1);
  k_gemm<<<g1, 256, 0, stream>>>(h116, w2t, b2, ffnf, nullptr, MS_, D_, F_, 0);
  k_add_ln<<<MS_, 256, 0, stream>>>(out1f, ffnf, ln2a, ln2b, out, nullptr);
}
